// PairformerBlock_58213986730473
// MI455X (gfx1250) — matrix operand order repaired and checked by reading, compiled, not run on hardware
//
#include <hip/hip_runtime.h>
#include <hip/hip_bf16.h>

// ---------------------------------------------------------------------------
// Pairformer block for MI455X (gfx1250, wave32, WMMA).
// All matmuls via v_wmma_f32_16x16x32_bf16 (bf16 A/B, f32 accumulate).
// ---------------------------------------------------------------------------

typedef __attribute__((ext_vector_type(16))) __bf16 v16bf;
typedef __attribute__((ext_vector_type(8)))  __bf16 v8bf;
typedef __attribute__((ext_vector_type(2)))  __bf16 v2bf;
typedef __attribute__((ext_vector_type(8)))  float  v8f;

#define BB 2
#define NN 384
#define DT 256
#define DP 128
#define DC 256
#define HH 8
#define DH 32
#define DFF 688
#define DFFP 704          // DFF padded to multiple of 32 for WMMA K loop
#define DFF2P 1408        // 2*DFF padded to multiple of 64 for full N-strips
#define DPH 512
#define EPSF 1e-5f

#if defined(__has_builtin)
#  if __has_builtin(__builtin_amdgcn_cvt_pk_bf16_f32)
#    define HAVE_CVT_PK_BF16 1
#  endif
#  if __has_builtin(__builtin_amdgcn_perm)
#    define HAVE_PERM 1
#  endif
#endif

// packed fp32 -> bf16; hardware cvt_pk when available, else round-half-up
// (+0x8000) and a single v_perm_b32 to pack the two high halves:
// 3 VALU ops per 2 elements.
__device__ __forceinline__ v2bf f2bf2(float lo, float hi) {
#ifdef HAVE_CVT_PK_BF16
    auto r = __builtin_amdgcn_cvt_pk_bf16_f32(lo, hi);
    return __builtin_bit_cast(v2bf, r);
#else
    unsigned ulo = __builtin_bit_cast(unsigned, lo) + 0x8000u;
    unsigned uhi = __builtin_bit_cast(unsigned, hi) + 0x8000u;
#  ifdef HAVE_PERM
    unsigned packed = __builtin_amdgcn_perm(uhi, ulo, 0x07060302u);
#  else
    unsigned packed = (ulo >> 16) | (uhi & 0xFFFF0000u);
#  endif
    return __builtin_bit_cast(v2bf, packed);
#endif
}
__device__ __forceinline__ __bf16 f2bf(float f) { return f2bf2(f, 0.0f)[0]; }

// A fragment (16x32 bf16, MxK): lane r=lane&15 is row, half=lane>>4.
// K(e) = ko + e + 8*half + (e>=8 ? 8 : 0)  -> two contiguous 16B loads.
__device__ __forceinline__ v16bf frag_a(const __bf16* A, int lda, int mo, int ko, int lane) {
    int r = lane & 15, half = lane >> 4;
    const __bf16* p = A + (size_t)(mo + r) * lda + ko + 8 * half;
    v8bf x0 = *(const v8bf*)p;
    v8bf x1 = *(const v8bf*)(p + 16);
    v16bf out;
#pragma unroll
    for (int i = 0; i < 8; i++) { out[i] = x0[i]; out[i + 8] = x1[i]; }
    return out;
}

// B fragment (32x16 bf16, KxN) from row-major W[Nc,K] (computing A·Wᵀ):
// lane n=lane&15 is output col (W row), K(e) = ko + e + 8*half + (e>=8 ? 8 : 0).
// Caller guarantees rows no..no+15 are valid (buffers padded to x16 rows).
__device__ __forceinline__ v16bf frag_b(const __bf16* B, int ldb, int no, int ko, int lane) {
    int n = lane & 15, half = lane >> 4;
    const __bf16* p = B + (size_t)(no + n) * ldb + ko + 8 * half;
    v8bf x0 = *(const v8bf*)p;
    v8bf x1 = *(const v8bf*)(p + 16);
    v16bf out;
#pragma unroll
    for (int i = 0; i < 8; i++) { out[i] = x0[i]; out[i + 8] = x1[i]; }
    return out;
}

// ---------------------------------------------------------------------------
// Generic batched GEMM: C[z] = (acc ? C[z] : 0) + A[z] · B[z]ᵀ
// block = 256 threads = 8 waves; wave owns a 16 x (NT*16) strip with NT
// accumulators (NT WMMAs per A-fragment load).  Every active wave owns a
// FULL strip (caller pads Nc accordingly); partial waves exit via a scalar
// branch (wave id is readfirstlane'd -> SGPR, no EXEC games around WMMA).
// C batch offset is 2-D: (z/batchH)*cOuter + (z%batchH)*cInner.
// Requires: M % 16 == 0, K % 32 == 0, strips fully valid or fully invalid.
// ---------------------------------------------------------------------------
template <int NT>
__global__ __launch_bounds__(256) void gemm_bf16_kernel(
    const __bf16* __restrict__ A, const __bf16* __restrict__ Bm, float* __restrict__ C,
    int M, int Nc, int K, int lda, int ldb, int ldc,
    long long aBatch, long long bBatch, int batchH,
    long long cOuter, long long cInner, int accumulate)
{
    int z = blockIdx.z;
    A  += (size_t)z * aBatch;
    Bm += (size_t)z * bBatch;
    C  += (size_t)(z / batchH) * cOuter + (size_t)(z % batchH) * cInner;

    int w = __builtin_amdgcn_readfirstlane(threadIdx.x >> 5);   // scalar wave id
    int lane = threadIdx.x & 31;
    int mo = blockIdx.y * 32 + (w >> 2) * 16;
    int noBase = blockIdx.x * (4 * NT * 16) + (w & 3) * (NT * 16);
    if (mo >= M || noBase >= Nc) return;                        // scalar branch

    int n = lane & 15, half = lane >> 4;
    v8f c[NT];
#pragma unroll
    for (int t = 0; t < NT; t++)
#pragma unroll
        for (int e = 0; e < 8; e++) c[t][e] = 0.0f;

    if (accumulate) {
#pragma unroll
        for (int t = 0; t < NT; t++)
#pragma unroll
            for (int e = 0; e < 8; e++)
                c[t][e] = C[(size_t)(mo + e + 8 * half) * ldc + noBase + t * 16 + n];
    }

    for (int ko = 0; ko < K; ko += 32) {
        if (ko + 32 < K) {
            __builtin_prefetch(A  + (size_t)(mo + n) * lda + ko + 32, 0, 3);
            __builtin_prefetch(Bm + (size_t)(noBase + n) * ldb + ko + 32, 0, 3);
        }
        v16bf a = frag_a(A, lda, mo, ko, lane);
#pragma unroll
        for (int t = 0; t < NT; t++) {
            v16bf b = frag_b(Bm, ldb, noBase + t * 16, ko, lane);
            c[t] = __builtin_amdgcn_wmma_f32_16x16x32_bf16(
                       false, a, false, b, (short)0, c[t], false, false);
        }
    }

#pragma unroll
    for (int t = 0; t < NT; t++)
#pragma unroll
        for (int e = 0; e < 8; e++)
            C[(size_t)(mo + e + 8 * half) * ldc + noBase + t * 16 + n] = c[t][e];
}

// ---------------------------------------------------------------------------
// Weight conversion kernels
// ---------------------------------------------------------------------------
// n must be even; 2 elements per thread, packed cvt + 4B store.
__global__ void cvt_bf16_kernel(const float* __restrict__ s, __bf16* __restrict__ d, int n) {
    int i = (blockIdx.x * 256 + threadIdx.x) * 2;
    if (i < n) *(v2bf*)(d + i) = f2bf2(s[i], s[i + 1]);
}
// Pad rows and/or cols with zeros (dst[dstRows, dstCols] from src[srcRows, srcCols]).
__global__ void cvt_pad2d_kernel(const float* __restrict__ s, __bf16* __restrict__ d,
                                 int srcRows, int srcCols, int dstRows, int dstCols) {
    int i = blockIdx.x * 256 + threadIdx.x;
    if (i < dstRows * dstCols) {
        int r = i / dstCols, c = i % dstCols;
        float v = (r < srcRows && c < srcCols) ? s[(size_t)r * srcCols + c] : 0.0f;
        d[i] = f2bf(v);
    }
}

// ---------------------------------------------------------------------------
// AdaLN modulation: mod[b,768] = cond @ ada_w.T + ada_b  (grid = (B, 2))
// ---------------------------------------------------------------------------
__global__ __launch_bounds__(256) void mod_kernel(
    const float* __restrict__ cond,
    const float* __restrict__ w1, const float* __restrict__ b1,
    const float* __restrict__ w2, const float* __restrict__ b2,
    float* __restrict__ mod1, float* __restrict__ mod2)
{
    int b = blockIdx.x, sel = blockIdx.y;
    const float* w  = sel ? w2 : w1;
    const float* bb = sel ? b2 : b1;
    float* out = sel ? mod2 : mod1;
    const float* cr = cond + (size_t)b * DC;
    for (int j = threadIdx.x; j < 3 * DT; j += 256) {
        float acc = bb[j];
        const float* wr = w + (size_t)j * DC;
        for (int i = 0; i < DC; i++) acc += wr[i] * cr[i];
        out[(size_t)b * 3 * DT + j] = acc;
    }
}

// h = LN(x)*(1+scale)+shift -> bf16    (grid = B*N rows, block = 256)
__global__ __launch_bounds__(256) void adaln_kernel(
    const float* __restrict__ x, const float* __restrict__ mod, __bf16* __restrict__ hb)
{
    int row = blockIdx.x, b = row / NN, tid = threadIdx.x;
    size_t off = (size_t)row * DT + tid;
    float t = x[off];
    __shared__ float r1[256], r2[256];
    r1[tid] = t; r2[tid] = t * t; __syncthreads();
    for (int st = 128; st > 0; st >>= 1) {
        if (tid < st) { r1[tid] += r1[tid + st]; r2[tid] += r2[tid + st]; }
        __syncthreads();
    }
    float mean = r1[0] * (1.0f / DT);
    float var  = r2[0] * (1.0f / DT) - mean * mean;
    float rs = rsqrtf(var + EPSF);
    float sc = mod[(size_t)b * 768 + 256 + tid];
    float sh = mod[(size_t)b * 768 + tid];
    hb[off] = f2bf((t - mean) * rs * (1.0f + sc) + sh);
}

// outTok = base + gate*delta ; optionally h2 = LN(outTok)*(1+scale2)+shift2
__global__ __launch_bounds__(256) void residual_adaln_kernel(
    const float* __restrict__ basex, const float* __restrict__ delta,
    const float* __restrict__ gateMod, const float* __restrict__ lnMod,
    float* __restrict__ outTok, __bf16* __restrict__ hb, int doLN)
{
    int row = blockIdx.x, b = row / NN, tid = threadIdx.x;
    size_t off = (size_t)row * DT + tid;
    float gate = gateMod[(size_t)b * 768 + 512 + tid];
    float t = basex[off] + gate * delta[off];
    outTok[off] = t;
    if (doLN) {
        __shared__ float r1[256], r2[256];
        r1[tid] = t; r2[tid] = t * t; __syncthreads();
        for (int st = 128; st > 0; st >>= 1) {
            if (tid < st) { r1[tid] += r1[tid + st]; r2[tid] += r2[tid + st]; }
            __syncthreads();
        }
        float mean = r1[0] * (1.0f / DT);
        float var  = r2[0] * (1.0f / DT) - mean * mean;
        float rs = rsqrtf(var + EPSF);
        float sc = lnMod[(size_t)b * 768 + 256 + tid];
        float sh = lnMod[(size_t)b * 768 + tid];
        hb[off] = f2bf((t - mean) * rs * (1.0f + sc) + sh);
    }
}

// Per-head LN of q/k (affine) + 1/sqrt(DH) folded into q; v -> transposed bf16.
// grid = B*N, block = 256 (one wave per head, DH=32 lanes).
__global__ __launch_bounds__(256) void qkln_kernel(
    const float* __restrict__ Qf, const float* __restrict__ Kf, const float* __restrict__ Vf,
    const float* __restrict__ qg, const float* __restrict__ qb_,
    const float* __restrict__ kg, const float* __restrict__ kb_,
    __bf16* __restrict__ qo, __bf16* __restrict__ ko_, __bf16* __restrict__ vT)
{
    int row = blockIdx.x, b = row / NN, nI = row % NN;
    int tid = threadIdx.x, h = tid >> 5, d = tid & 31;
    size_t off = (size_t)row * DT + tid;
    float q = Qf[off], k = Kf[off], v = Vf[off];

    auto lnf = [&](float x, const float* gg, const float* bbv) -> float {
        float s = x, ss = x * x;
        for (int m = 1; m < 32; m <<= 1) { s += __shfl_xor(s, m, 32); ss += __shfl_xor(ss, m, 32); }
        float mean = s * (1.0f / DH), var = ss * (1.0f / DH) - mean * mean;
        return (x - mean) * rsqrtf(var + EPSF) * gg[d] + bbv[d];
    };
    float qn = lnf(q, qg, qb_) * 0.17677669529663687f;   // 1/sqrt(32)
    float kn = lnf(k, kg, kb_);

    size_t qidx = (((size_t)(b * HH + h) * NN) + nI) * DH + d;
    qo[qidx]  = f2bf(qn);
    ko_[qidx] = f2bf(kn);
    vT[(((size_t)(b * HH + h) * DH) + d) * NN + nI] = f2bf(v);
}

// Pair bias: scores[b,h,q,k] = sum_p pair[b,q,k,p] * Wpb[h,p]
// Wpb padded to 16 rows of zeros -> no guard in the WMMA loop.
// A fragments converted f32->bf16 with packed cvt (3 ops / 2 elems).
// block = 256 (8 waves), wave handles 16 pair positions. grid = B*N*N/128.
__global__ __launch_bounds__(256) void pair_bias_kernel(
    const float* __restrict__ pairIn, const __bf16* __restrict__ wpbb, float* __restrict__ S)
{
    int w = __builtin_amdgcn_readfirstlane(threadIdx.x >> 5);
    int lane = threadIdx.x & 31;
    int n = lane & 15, half = lane >> 4;
    size_t mo = ((size_t)blockIdx.x * 8 + w) * 16;

    v8f c;
#pragma unroll
    for (int e = 0; e < 8; e++) c[e] = 0.0f;

#pragma unroll
    for (int ko = 0; ko < DP; ko += 32) {
        const float* p0 = pairIn + (mo + n) * (size_t)DP + ko + 8 * half;
        v16bf a;
#pragma unroll
        for (int i = 0; i < 8; i += 2) {
            v2bf t0 = f2bf2(p0[i],      p0[i + 1]);
            v2bf t1 = f2bf2(p0[16 + i], p0[17 + i]);
            a[i] = t0[0];      a[i + 1] = t0[1];
            a[8 + i] = t1[0];  a[9 + i] = t1[1];
        }
        v16bf b = frag_b(wpbb, DP, 0, ko, lane);
        c = __builtin_amdgcn_wmma_f32_16x16x32_bf16(false, a, false, b, (short)0, c, false, false);
    }

    if (n < HH) {
#pragma unroll
        for (int e = 0; e < 8; e++) {
            size_t pos = mo + e + 8 * half;
            int b = (int)(pos / ((size_t)NN * NN));
            size_t rem = pos % ((size_t)NN * NN);
            int q = (int)(rem / NN), kk = (int)(rem % NN);
            S[(((size_t)(b * HH + n) * NN) + q) * NN + kk] = c[e];
        }
    }
}

// Masked softmax over k (row = b*H*N + h*N + q), out bf16. block = 128, 3 cols/thread.
__global__ __launch_bounds__(128) void softmax_kernel(
    const float* __restrict__ S, const unsigned char* __restrict__ mask, __bf16* __restrict__ P)
{
    int row = blockIdx.x, b = row / (HH * NN), tid = threadIdx.x;
    const float* s = S + (size_t)row * NN;
    __bf16* p = P + (size_t)row * NN;
    const unsigned char* mk = mask + (size_t)b * NN;
    __shared__ float red[128];

    float v[3]; int ok[3];
    float mx = -3.0e38f;
#pragma unroll
    for (int i = 0; i < 3; i++) {
        int k = tid + i * 128;
        ok[i] = mk[k] != 0;
        v[i] = ok[i] ? s[k] : -3.0e38f;
        mx = fmaxf(mx, v[i]);
    }
    red[tid] = mx; __syncthreads();
    for (int st = 64; st > 0; st >>= 1) { if (tid < st) red[tid] = fmaxf(red[tid], red[tid + st]); __syncthreads(); }
    mx = red[0]; __syncthreads();

    float e[3], sum = 0.0f;
#pragma unroll
    for (int i = 0; i < 3; i++) { e[i] = ok[i] ? __expf(v[i] - mx) : 0.0f; sum += e[i]; }
    red[tid] = sum; __syncthreads();
    for (int st = 64; st > 0; st >>= 1) { if (tid < st) red[tid] += red[tid + st]; __syncthreads(); }
    sum = red[0];
    float inv = (sum > 0.0f) ? 1.0f / sum : 0.0f;
#pragma unroll
    for (int i = 0; i < 3; i++) p[tid + i * 128] = f2bf(e[i] * inv);
}

// SwiGLU: s = silu(mid[:, :688]) * mid[:, 688:1376], mid stride DFF2P,
// output zero-padded to 704 cols (bf16).
__global__ __launch_bounds__(256) void swiglu_kernel(
    const float* __restrict__ mid, __bf16* __restrict__ sb)
{
    int r = blockIdx.x;
    const float* m = mid + (size_t)r * DFF2P;
    __bf16* o = sb + (size_t)r * DFFP;
    for (int c = threadIdx.x; c < DFFP; c += 256) {
        float val = 0.0f;
        if (c < DFF) {
            float a = m[c], g = m[DFF + c];
            val = (a / (1.0f + __expf(-a))) * g;
        }
        o[c] = f2bf(val);
    }
}

// ---------------------------------------------------------------------------
// Fused pair transition: LN -> (128->512) GEMM + SiLU -> (512->128) GEMM
// + bias + residual.  256 threads / 8 waves per 32 pair rows, LDS-staged.
// K-loop hoisted outside N-tiles: one A fragment feeds 8 (phase B) or 2
// (phase C) WMMAs.  Wave id is scalar (readfirstlane).
// ---------------------------------------------------------------------------
__global__ __launch_bounds__(256) void pair_transition_kernel(
    const float* __restrict__ pairIn,
    const float* __restrict__ lnG, const float* __restrict__ lnB,
    const __bf16* __restrict__ w1b, const float* __restrict__ b1,
    const __bf16* __restrict__ w2b, const float* __restrict__ b2,
    float* __restrict__ pairOut)
{
    __shared__ __bf16 aT[32 * DP];    //  8 KB LN'd activations
    __shared__ __bf16 hT[32 * DPH];   // 32 KB SiLU hidden
    int tid = threadIdx.x;
    size_t base = (size_t)blockIdx.x * 32;

    // ---- Phase A: LayerNorm of 32 rows x 128 cols, 8 lanes per row ----
    {
        int r = tid >> 3;                 // 0..31
        int cg = (tid & 7) * 16;          // 16 cols per lane
        const float* xr = pairIn + (base + r) * (size_t)DP + cg;
        float vals[16], s = 0.0f, ss = 0.0f;
#pragma unroll
        for (int i = 0; i < 16; i++) { float x = xr[i]; vals[i] = x; s += x; ss += x * x; }
        for (int m = 1; m < 8; m <<= 1) { s += __shfl_xor(s, m, 32); ss += __shfl_xor(ss, m, 32); }
        float mean = s * (1.0f / DP);
        float var  = ss * (1.0f / DP) - mean * mean;
        float rs = rsqrtf(var + EPSF);
#pragma unroll
        for (int i = 0; i < 16; i += 2) {
            int c = cg + i;
            float y0 = (vals[i]     - mean) * rs * lnG[c]     + lnB[c];
            float y1 = (vals[i + 1] - mean) * rs * lnG[c + 1] + lnB[c + 1];
            *(v2bf*)(&aT[r * DP + c]) = f2bf2(y0, y1);
        }
    }
    __syncthreads();

    int w = __builtin_amdgcn_readfirstlane(tid >> 5);   // scalar wave id
    int lane = tid & 31;
    int n16 = lane & 15, half = lane >> 4;
    int mt = w >> 2;                      // M-tile 0..1 (scalar)
    int quarter = w & 3;                  // 128-col quarter of hidden (scalar)

    // ---- Phase B: hidden = SiLU(aT · W1ᵀ + b1) ; 8 acc, A reused 8x ----
    {
        v8f acc[8];
#pragma unroll
        for (int t = 0; t < 8; t++)
#pragma unroll
            for (int e = 0; e < 8; e++) acc[t][e] = 0.0f;

#pragma unroll
        for (int ko = 0; ko < DP; ko += 32) {
            v16bf a = frag_a(aT, DP, mt * 16, ko, lane);
#pragma unroll
            for (int t = 0; t < 8; t++) {
                v16bf b = frag_b(w1b, DP, quarter * 128 + t * 16, ko, lane);
                acc[t] = __builtin_amdgcn_wmma_f32_16x16x32_bf16(
                             false, a, false, b, (short)0, acc[t], false, false);
            }
        }
#pragma unroll
        for (int t = 0; t < 8; t++) {
            int no = quarter * 128 + t * 16;
            float bias1 = b1[no + n16];
#pragma unroll
            for (int e = 0; e < 8; e++) {
                float x = acc[t][e] + bias1;
                float sv = x / (1.0f + __expf(-x));
                hT[(mt * 16 + e + 8 * half) * DPH + no + n16] = f2bf(sv);
            }
        }
    }
    __syncthreads();

    // ---- Phase C: out = pair + hidden · W2ᵀ + b2 ; 2 acc, A reused 2x ----
    {
        int no0 = (quarter * 2) * 16;
        int no1 = (quarter * 2 + 1) * 16;
        v8f acc0, acc1;
#pragma unroll
        for (int e = 0; e < 8; e++) { acc0[e] = 0.0f; acc1[e] = 0.0f; }

        for (int ko = 0; ko < DPH; ko += 32) {
            v16bf a = frag_a(hT, DPH, mt * 16, ko, lane);
            v16bf b0 = frag_b(w2b, DPH, no0, ko, lane);
            acc0 = __builtin_amdgcn_wmma_f32_16x16x32_bf16(
                       false, a, false, b0, (short)0, acc0, false, false);
            v16bf b1f = frag_b(w2b, DPH, no1, ko, lane);
            acc1 = __builtin_amdgcn_wmma_f32_16x16x32_bf16(
                       false, a, false, b1f, (short)0, acc1, false, false);
        }

        float bias20 = b2[no0 + n16];
        float bias21 = b2[no1 + n16];
#pragma unroll
        for (int e = 0; e < 8; e++) {
            size_t row = base + mt * 16 + e + 8 * half;
            size_t i0 = row * (size_t)DP + no0 + n16;
            size_t i1 = row * (size_t)DP + no1 + n16;
            pairOut[i0] = pairIn[i0] + acc0[e] + bias20;
            pairOut[i1] = pairIn[i1] + acc1[e] + bias21;
        }
    }
}

// ---------------------------------------------------------------------------
// Host launcher
// ---------------------------------------------------------------------------
extern "C" void kernel_launch(void* const* d_in, const int* in_sizes, int n_in,
                              void* d_out, int out_size, void* d_ws, size_t ws_size,
                              hipStream_t stream)
{
    (void)in_sizes; (void)n_in; (void)out_size; (void)ws_size;

    const float* token  = (const float*)d_in[0];
    const float* pairIn = (const float*)d_in[1];
    const float* cond   = (const float*)d_in[2];
    const unsigned char* mask = (const unsigned char*)d_in[3];
    const float* Wq = (const float*)d_in[4];
    const float* Wk = (const float*)d_in[5];
    const float* Wv = (const float*)d_in[6];
    const float* Wo = (const float*)d_in[7];
    const float* qn_g = (const float*)d_in[8];
    const float* qn_b = (const float*)d_in[9];
    const float* kn_g = (const float*)d_in[10];
    const float* kn_b = (const float*)d_in[11];
    const float* Wpb  = (const float*)d_in[12];
    const float* ada1_w = (const float*)d_in[13];
    const float* ada1_b = (const float*)d_in[14];
    const float* ada2_w = (const float*)d_in[15];
    const float* ada2_b = (const float*)d_in[16];
    const float* ffn_w1 = (const float*)d_in[17];
    const float* ffn_w2 = (const float*)d_in[18];
    const float* pt_ln_g = (const float*)d_in[19];
    const float* pt_ln_b = (const float*)d_in[20];
    const float* pt_w1 = (const float*)d_in[21];
    const float* pt_b1 = (const float*)d_in[22];
    const float* pt_w2 = (const float*)d_in[23];
    const float* pt_b2 = (const float*)d_in[24];

    float* outTok  = (float*)d_out;                       // [B,N,DT]
    float* outPair = outTok + (size_t)BB * NN * DT;       // [B,N,N,DP]

    // workspace bump allocator (256B aligned segments)
    char* wp = (char*)d_ws;
    auto alloc = [&](size_t bytes) -> void* {
        void* r = (void*)wp;
        wp += (bytes + 255) & ~(size_t)255;
        return r;
    };
    const size_t TOK = (size_t)BB * NN * DT;              // 196608
    const size_t SCR = (size_t)BB * HH * NN * NN;         // 2359296

    __bf16* wqb   = (__bf16*)alloc(DT * DT * 2);
    __bf16* wkb   = (__bf16*)alloc(DT * DT * 2);
    __bf16* wvb   = (__bf16*)alloc(DT * DT * 2);
    __bf16* wob   = (__bf16*)alloc(DT * DT * 2);
    __bf16* ffn1b = (__bf16*)alloc((size_t)DFF2P * DT * 2);   // rows padded 1376->1408
    __bf16* ffn2b = (__bf16*)alloc((size_t)DT * DFFP * 2);    // cols padded 688->704
    __bf16* ptw1b = (__bf16*)alloc((size_t)DPH * DP * 2);
    __bf16* ptw2b = (__bf16*)alloc((size_t)DP * DPH * 2);
    __bf16* wpbb  = (__bf16*)alloc((size_t)16 * DP * 2);      // rows padded 8->16
    float*  mod1  = (float*)alloc(BB * 3 * DT * 4);
    float*  mod2  = (float*)alloc(BB * 3 * DT * 4);
    __bf16* h1b   = (__bf16*)alloc(TOK * 2);
    float*  Qf    = (float*)alloc(TOK * 4);
    float*  Kf    = (float*)alloc(TOK * 4);
    float*  Vf    = (float*)alloc(TOK * 4);
    __bf16* qb    = (__bf16*)alloc(TOK * 2);
    __bf16* kb    = (__bf16*)alloc(TOK * 2);
    __bf16* vT    = (__bf16*)alloc(TOK * 2);
    float*  scores= (float*)alloc(SCR * 4);
    __bf16* pb    = (__bf16*)alloc(SCR * 2);
    float*  attnO = (float*)alloc(TOK * 4);
    __bf16* attnOb= (__bf16*)alloc(TOK * 2);
    float*  attnW = (float*)alloc(TOK * 4);
    __bf16* h2b   = (__bf16*)alloc(TOK * 2);
    float*  ffnmid= (float*)alloc((size_t)BB * NN * DFF2P * 4);
    __bf16* sb    = (__bf16*)alloc((size_t)BB * NN * DFFP * 2);
    float*  ffnout= (float*)alloc(TOK * 4);

    auto cvt = [&](const float* s, __bf16* d, int n) {
        cvt_bf16_kernel<<<dim3((n / 2 + 255) / 256), dim3(256), 0, stream>>>(s, d, n);
    };
    auto gemm4 = [&](const __bf16* A, const __bf16* Bm, float* C, int M, int Nc, int K,
                     int lda, int ldb, int ldc, int batch, long long aB, long long bB,
                     int bH, long long cO, long long cI, int acc) {
        dim3 g((Nc + 255) / 256, (M + 31) / 32, batch);
        gemm_bf16_kernel<4><<<g, dim3(256), 0, stream>>>(A, Bm, C, M, Nc, K, lda, ldb, ldc,
                                                         aB, bB, bH, cO, cI, acc);
    };
    auto gemm2 = [&](const __bf16* A, const __bf16* Bm, float* C, int M, int Nc, int K,
                     int lda, int ldb, int ldc, int batch, long long aB, long long bB,
                     int bH, long long cO, long long cI, int acc) {
        dim3 g((Nc + 127) / 128, (M + 31) / 32, batch);
        gemm_bf16_kernel<2><<<g, dim3(256), 0, stream>>>(A, Bm, C, M, Nc, K, lda, ldb, ldc,
                                                         aB, bB, bH, cO, cI, acc);
    };

    // 1) weight conversion (row/col padded where WMMA tiling requires it)
    cvt(Wq, wqb, DT * DT);  cvt(Wk, wkb, DT * DT);
    cvt(Wv, wvb, DT * DT);  cvt(Wo, wob, DT * DT);
    cvt(pt_w1, ptw1b, DPH * DP);
    cvt(pt_w2, ptw2b, DP * DPH);
    cvt_pad2d_kernel<<<dim3((DFF2P * DT + 255) / 256), dim3(256), 0, stream>>>(
        ffn_w1, ffn1b, 2 * DFF, DT, DFF2P, DT);
    cvt_pad2d_kernel<<<dim3((DT * DFFP + 255) / 256), dim3(256), 0, stream>>>(
        ffn_w2, ffn2b, DT, DFF, DT, DFFP);
    cvt_pad2d_kernel<<<dim3((16 * DP + 255) / 256), dim3(256), 0, stream>>>(
        Wpb, wpbb, HH, DP, 16, DP);

    // 2) AdaLN modulation for both sublayers
    mod_kernel<<<dim3(BB, 2), dim3(256), 0, stream>>>(cond, ada1_w, ada1_b, ada2_w, ada2_b, mod1, mod2);

    // 3) h1 = AdaLN(token, mod1)
    adaln_kernel<<<dim3(BB * NN), dim3(256), 0, stream>>>(token, mod1, h1b);

    // 4) QKV projections (WMMA)
    gemm4(h1b, wqb, Qf, BB * NN, DT, DT, DT, DT, DT, 1, 0, 0, 1, 0, 0, 0);
    gemm4(h1b, wkb, Kf, BB * NN, DT, DT, DT, DT, DT, 1, 0, 0, 1, 0, 0, 0);
    gemm4(h1b, wvb, Vf, BB * NN, DT, DT, DT, DT, DT, 1, 0, 0, 1, 0, 0, 0);

    // 5) per-head q/k LN + layout
    qkln_kernel<<<dim3(BB * NN), dim3(256), 0, stream>>>(Qf, Kf, Vf, qn_g, qn_b, kn_g, kn_b, qb, kb, vT);

    // 6) pair bias into scores, then scores += q·kᵀ (batched over b,h)
    pair_bias_kernel<<<dim3(BB * NN * NN / 128), dim3(256), 0, stream>>>(pairIn, wpbb, scores);
    gemm4(qb, kb, scores, NN, NN, DH, DH, DH, NN,
          BB * HH, (long long)NN * DH, (long long)NN * DH,
          HH, (long long)HH * NN * NN, (long long)NN * NN, 1);

    // 7) masked softmax -> bf16 probabilities
    softmax_kernel<<<dim3(BB * HH * NN), dim3(128), 0, stream>>>(scores, mask, pb);

    // 8) attnO[b,q,h,d] = P · V  (batched, Nc=32 -> NT=2; C offset 2-D in (b,h))
    gemm2(pb, vT, attnO, NN, DH, NN, NN, NN, DT,
          BB * HH, (long long)NN * NN, (long long)DH * NN,
          HH, (long long)NN * DT, (long long)DH, 0);

    // 9) output projection + gated residual + AdaLN2
    cvt(attnO, attnOb, (int)TOK);
    gemm4(attnOb, wob, attnW, BB * NN, DT, DT, DT, DT, DT, 1, 0, 0, 1, 0, 0, 0);
    residual_adaln_kernel<<<dim3(BB * NN), dim3(256), 0, stream>>>(
        token, attnW, mod1, mod2, outTok, h2b, 1);

    // 10) FFN: mid = h2 · W1ᵀ (N padded to 1408) ; SwiGLU ; out = s · W2ᵀ ; residual
    gemm4(h2b, ffn1b, ffnmid, BB * NN, DFF2P, DT, DT, DT, DFF2P, 1, 0, 0, 1, 0, 0, 0);
    swiglu_kernel<<<dim3(BB * NN), dim3(256), 0, stream>>>(ffnmid, sb);
    gemm4(sb, ffn2b, ffnout, BB * NN, DT, DFFP, DFFP, DFFP, DT, 1, 0, 0, 1, 0, 0, 0);
    residual_adaln_kernel<<<dim3(BB * NN), dim3(256), 0, stream>>>(
        outTok, ffnout, mod2, mod2, outTok, h2b, 0);

    // 11) pair transition (dominant compute, fully fused WMMA)
    pair_transition_kernel<<<dim3(BB * NN * NN / 32), dim3(256), 0, stream>>>(
        pairIn, pt_ln_g, pt_ln_b, ptw1b, pt_b1, ptw2b, pt_b2, outPair);
}
